// StandardAttention_78245714199357
// MI455X (gfx1250) — compile-verified, matrix operand order checked by reading, not run on hardware
//
#include <hip/hip_runtime.h>
#include <hip/hip_bf16.h>

// ---------------------------------------------------------------------------
// MI455X (gfx1250) causal MHA, bf16 WMMA end-to-end.
// B=2, T=2048, C=1024, H=16, D=64.  ~69 GFLOP, working set << 192MB L2 ->
// compute bound -> all GEMMs on v_wmma_f32_16x16x32_bf16; softmax reductions
// on pure-VALU DPP16 butterflies (co-executes with the XDL pipe).
// ---------------------------------------------------------------------------

#define Bc 2
#define Tc 2048
#define Cc 1024
#define Hc 16
#define Dc 64
#define SCALEc 0.125f           // 1/sqrt(64)
#define ROWS_TOTAL (Bc*Hc*Tc)   // 65536 softmax rows

typedef __attribute__((ext_vector_type(16))) __bf16 v16bf;
typedef __attribute__((ext_vector_type(8)))  float  v8f;

__device__ __forceinline__ unsigned short f2bf(float f) {
    unsigned int u = __float_as_uint(f);
    u += 0x7FFFu + ((u >> 16) & 1u);        // round-to-nearest-even
    return (unsigned short)(u >> 16);
}

__device__ __forceinline__ v8f wmma_bf16(v16bf a, v16bf b, v8f c) {
    return __builtin_amdgcn_wmma_f32_16x16x32_bf16(
        false, a, false, b, (short)0, c, false, false);
}

// A/B fragment loader for 16x16x32 bf16 WMMA (ISA 7.12.2 layout):
// lane&15 selects the M row (A) / N column (B); K-chunks {kb..kb+7, kb+16..kb+23},
// kb = (lane>>4)*8.  Both chunks are 16B-contiguous in memory by construction.
__device__ __forceinline__ v16bf load_frag(const unsigned short* __restrict__ base,
                                           int stride, int rowBase, int colBase, int lane) {
    const unsigned short* p = base + (long)(rowBase + (lane & 15)) * stride
                                   + colBase + ((lane >> 4) << 3);
    v16bf r;
    uint4* pr = reinterpret_cast<uint4*>(&r);
    pr[0] = *reinterpret_cast<const uint4*>(p);
    pr[1] = *reinterpret_cast<const uint4*>(p + 16);
    return r;
}

// ---------------------------------------------------------------------------
// Pure-VALU 16-lane butterfly reductions via DPP16 (no LDS, no DScnt waits).
// Steps: xor1 = quad_perm[1,0,3,2] (0xB1), xor2 = quad_perm[2,3,0,1] (0x4E),
// then row_half_mirror (0x141) and row_mirror (0x140).  All patterns are
// row-local (16 lanes), so lane groups 0-15 / 16-31 reduce independently --
// exactly matching the WMMA C-tile row mapping on wave32.
// ---------------------------------------------------------------------------
__device__ __forceinline__ float dpp_bfly_max(float v) {
    float o;
    o = __int_as_float(__builtin_amdgcn_update_dpp(0, __float_as_int(v), 0xB1,  0xF, 0xF, true));
    v = fmaxf(v, o);
    o = __int_as_float(__builtin_amdgcn_update_dpp(0, __float_as_int(v), 0x4E,  0xF, 0xF, true));
    v = fmaxf(v, o);
    o = __int_as_float(__builtin_amdgcn_update_dpp(0, __float_as_int(v), 0x141, 0xF, 0xF, true));
    v = fmaxf(v, o);
    o = __int_as_float(__builtin_amdgcn_update_dpp(0, __float_as_int(v), 0x140, 0xF, 0xF, true));
    return fmaxf(v, o);
}
__device__ __forceinline__ float dpp_bfly_sum(float v) {
    float o;
    o = __int_as_float(__builtin_amdgcn_update_dpp(0, __float_as_int(v), 0xB1,  0xF, 0xF, true));
    v += o;
    o = __int_as_float(__builtin_amdgcn_update_dpp(0, __float_as_int(v), 0x4E,  0xF, 0xF, true));
    v += o;
    o = __int_as_float(__builtin_amdgcn_update_dpp(0, __float_as_int(v), 0x141, 0xF, 0xF, true));
    v += o;
    o = __int_as_float(__builtin_amdgcn_update_dpp(0, __float_as_int(v), 0x140, 0xF, 0xF, true));
    return v + o;
}

// ---------------------------------------------------------------------------
// fp32 -> bf16 conversion
// ---------------------------------------------------------------------------
__global__ void f32_to_bf16_k(const float* __restrict__ in,
                              unsigned short* __restrict__ out, long n) {
    long i = (long)blockIdx.x * blockDim.x + threadIdx.x;
    if (i < n) out[i] = f2bf(in[i]);
}

__global__ void zero_ent_k(float* p) { *p = 0.0f; }

// ---------------------------------------------------------------------------
// Tiled WMMA GEMM:  Y[M,N] = A[M,K](bf16) @ W[N,K]^T(bf16) + bias
// 256 threads = 8 waves; wave w -> 16x64 tile.  Software-pipelined k-loop:
// fragments for k+32 are issued before the WMMAs that consume k's fragments,
// so global_load_b128 latency overlaps the matrix pipe.
// MODE 0: store bf16 row-major            (Q, K projections)
// MODE 1: store bf16 as Vt[bh*64+d][T]    (V projection, per-head transpose)
// MODE 2: store f32 row-major             (final output projection)
// ---------------------------------------------------------------------------
template<int MODE>
__global__ __launch_bounds__(256) void gemm_bf16_k(
    const unsigned short* __restrict__ A,
    const unsigned short* __restrict__ W,
    const float* __restrict__ bias,
    void* __restrict__ out, int M, int N, int K)
{
    const int lane = threadIdx.x & 31;
    const int w    = threadIdx.x >> 5;
    const int m0   = blockIdx.y * 128 + w * 16;
    const int n0   = blockIdx.x * 64;

    v8f acc[4];
#pragma unroll
    for (int nt = 0; nt < 4; ++nt)
#pragma unroll
        for (int r = 0; r < 8; ++r) acc[nt][r] = 0.0f;

    // prologue: fragments for k0 = 0
    v16bf a_cur = load_frag(A, K, m0, 0, lane);
    v16bf b_cur[4];
#pragma unroll
    for (int nt = 0; nt < 4; ++nt) b_cur[nt] = load_frag(W, K, n0 + nt * 16, 0, lane);

    for (int k0 = 0; k0 < K; k0 += 32) {
        const int kn = (k0 + 32 < K) ? (k0 + 32) : 0;   // clamped (last iter redundant)
        v16bf a_nxt = load_frag(A, K, m0, kn, lane);
        v16bf b_nxt[4];
#pragma unroll
        for (int nt = 0; nt < 4; ++nt) b_nxt[nt] = load_frag(W, K, n0 + nt * 16, kn, lane);
#pragma unroll
        for (int nt = 0; nt < 4; ++nt) acc[nt] = wmma_bf16(a_cur, b_cur[nt], acc[nt]);
        a_cur = a_nxt;
#pragma unroll
        for (int nt = 0; nt < 4; ++nt) b_cur[nt] = b_nxt[nt];
    }

    const int half = lane >> 4;
    const int nl   = lane & 15;
#pragma unroll
    for (int nt = 0; nt < 4; ++nt) {
        const int n  = n0 + nt * 16 + nl;
        const float bb = bias[n];
#pragma unroll
        for (int r = 0; r < 8; ++r) {
            const int m = m0 + r + half * 8;
            const float v = acc[nt][r] + bb;
            if (MODE == 2) {
                reinterpret_cast<float*>(out)[(long)m * N + n] = v;
            } else if (MODE == 0) {
                reinterpret_cast<unsigned short*>(out)[(long)m * N + n] = f2bf(v);
            } else { // MODE 1: Vt[(b*H+h)*64 + d][t]
                const int b = m >> 11, t = m & (Tc - 1);
                const int h = n >> 6,  d = n & (Dc - 1);
                const long idx = ((long)((b * Hc + h) * Dc + d)) * Tc + t;
                reinterpret_cast<unsigned short*>(out)[idx] = f2bf(v);
            }
        }
    }
}

// ---------------------------------------------------------------------------
// Flash attention (causal) + online entropy.
// grid = (T/128, B*H); 8 waves/block, wave owns 16 queries.
// Per 32-key tile: all K and V fragments are issued up front (V loads are in
// flight through the whole softmax), 4 WMMA for QK^T, online softmax with
// running (m, l, e) using DPP butterflies, P staged through LDS (C-layout f32
// -> A-layout bf16), 4 WMMA for PV.  entropy_row = ln(l) - e/l.
// ---------------------------------------------------------------------------
__global__ __launch_bounds__(256) void flash_attn_k(
    const unsigned short* __restrict__ Qb,
    const unsigned short* __restrict__ Kb,
    const unsigned short* __restrict__ Vt,
    unsigned short* __restrict__ Ctx,
    float* __restrict__ entOut)
{
    __shared__ unsigned short Plds[8][16 * 32];   // 1KB per wave

    const int lane = threadIdx.x & 31;
    const int w    = threadIdx.x >> 5;
    const int bh   = blockIdx.y;
    const int b    = bh >> 4, h = bh & 15;
    const int q0   = blockIdx.x * 128 + w * 16;
    const int rowQ = b * Tc + q0;
    const int colH = h * Dc;
    const int half = lane >> 4;
    const int nl   = lane & 15;

    // Q fragments (d 0..31 and 32..63), reused across all key tiles.
    const v16bf aq0 = load_frag(Qb, Cc, rowQ, colH,      lane);
    const v16bf aq1 = load_frag(Qb, Cc, rowQ, colH + 32, lane);

    float m[8], l[8], e[8];
    v8f O[4];
#pragma unroll
    for (int r = 0; r < 8; ++r) { m[r] = -1e30f; l[r] = 0.0f; e[r] = 0.0f; }
#pragma unroll
    for (int nt = 0; nt < 4; ++nt)
#pragma unroll
        for (int r = 0; r < 8; ++r) O[nt][r] = 0.0f;

    for (int t0 = 0; t0 < q0 + 16; t0 += 32) {
        if (t0 + 32 < Tc)   // speculative prefetch of next K tile -> global_prefetch_b8
            __builtin_prefetch(Kb + (long)(b * Tc + t0 + 32 + nl) * Cc + colH, 0, 1);

        // ---- issue ALL tile loads up front ---------------------------------
        v16bf kf00 = load_frag(Kb, Cc, b * Tc + t0,      colH,      lane);
        v16bf kf01 = load_frag(Kb, Cc, b * Tc + t0,      colH + 32, lane);
        v16bf kf10 = load_frag(Kb, Cc, b * Tc + t0 + 16, colH,      lane);
        v16bf kf11 = load_frag(Kb, Cc, b * Tc + t0 + 16, colH + 32, lane);
        v16bf vf[4];
#pragma unroll
        for (int nt = 0; nt < 4; ++nt)
            vf[nt] = load_frag(Vt, Tc, bh * Dc + nt * 16, t0, lane);

        // ---- scores: two 16x16 tiles over 32 keys --------------------------
        v8f S0, S1;
#pragma unroll
        for (int r = 0; r < 8; ++r) { S0[r] = 0.0f; S1[r] = 0.0f; }
        S0 = wmma_bf16(aq0, kf00, S0);
        S0 = wmma_bf16(aq1, kf01, S0);
        S1 = wmma_bf16(aq0, kf10, S1);
        S1 = wmma_bf16(aq1, kf11, S1);

        // ---- online softmax + entropy statistic (pure-VALU DPP) ------------
        float p0[8], p1[8];
#pragma unroll
        for (int r = 0; r < 8; ++r) {
            const int q  = q0 + r + half * 8;
            const int k0 = t0 + nl;
            float s0 = S0[r] * SCALEc; if (k0 > q)       s0 = -1e30f;
            float s1 = S1[r] * SCALEc; if (k0 + 16 > q)  s1 = -1e30f;

            const float rmax  = dpp_bfly_max(fmaxf(s0, s1));
            const float mnew  = fmaxf(m[r], rmax);
            const float alpha = __expf(m[r] - mnew);
            e[r] = alpha * (e[r] + (m[r] - mnew) * l[r]);
            l[r] = alpha * l[r];

            const float e0 = __expf(s0 - mnew);
            const float e1 = __expf(s1 - mnew);
            l[r] += dpp_bfly_sum(e0 + e1);
            e[r] += dpp_bfly_sum(e0 * (s0 - mnew) + e1 * (s1 - mnew));
            m[r] = mnew;
            p0[r] = e0; p1[r] = e1;
#pragma unroll
            for (int nt = 0; nt < 4; ++nt) O[nt][r] *= alpha;
        }

        // ---- stage P through LDS: C-layout f32 -> row-major bf16 [16][32] --
#pragma unroll
        for (int r = 0; r < 8; ++r) {
            Plds[w][(half * 8 + r) * 32 + nl]      = f2bf(p0[r]);
            Plds[w][(half * 8 + r) * 32 + 16 + nl] = f2bf(p1[r]);
        }
        // same-wave LDS ops are in-order; fence compiler + drain DScnt once
        asm volatile("s_wait_dscnt 0x0" ::: "memory");

        // read back in A-fragment layout (ds_load_b128 x2)
        const unsigned short* lp = &Plds[w][nl * 32 + (half << 3)];
        v16bf pa;
        uint4* pp = reinterpret_cast<uint4*>(&pa);
        pp[0] = *reinterpret_cast<const uint4*>(lp);
        pp[1] = *reinterpret_cast<const uint4*>(lp + 16);
        asm volatile("" ::: "memory");

        // ---- PV: O(16x64) += P(16x32) @ V(32x64), V pre-transposed ----------
#pragma unroll
        for (int nt = 0; nt < 4; ++nt)
            O[nt] = wmma_bf16(pa, vf[nt], O[nt]);
    }

    // ---- epilogue: normalize, store context (bf16), accumulate entropy ----
    float entSum = 0.0f;
#pragma unroll
    for (int r = 0; r < 8; ++r) {
        const float invl = 1.0f / l[r];
        const long  qrow = (long)b * Tc + q0 + r + half * 8;
#pragma unroll
        for (int nt = 0; nt < 4; ++nt) {
            Ctx[qrow * Cc + colH + nt * 16 + nl] = f2bf(O[nt][r] * invl);
        }
        entSum += __logf(l[r]) - e[r] * invl;
    }
    if (nl == 0)   // lanes 0 and 16 each cover 8 rows
        atomicAdd(entOut, entSum * (1.0f / (float)ROWS_TOTAL));
}

// ---------------------------------------------------------------------------
// Host-side launcher
// ---------------------------------------------------------------------------
extern "C" void kernel_launch(void* const* d_in, const int* in_sizes, int n_in,
                              void* d_out, int out_size, void* d_ws, size_t ws_size,
                              hipStream_t stream) {
    (void)in_sizes; (void)n_in; (void)out_size; (void)ws_size;

    const float* x  = (const float*)d_in[0];
    // d_in[1] is the causal mask; causality is applied analytically.
    const float* wq = (const float*)d_in[2];
    const float* bq = (const float*)d_in[3];
    const float* wk = (const float*)d_in[4];
    const float* bk = (const float*)d_in[5];
    const float* wv = (const float*)d_in[6];
    const float* bv = (const float*)d_in[7];
    const float* wo = (const float*)d_in[8];
    const float* bo = (const float*)d_in[9];

    char* ws = (char*)d_ws;
    const size_t MB = 1u << 20;
    unsigned short* Xb  = (unsigned short*)(ws);             //  8 MB  [4096,1024] bf16
    unsigned short* Wqb = (unsigned short*)(ws +  8 * MB);   //  2 MB
    unsigned short* Wkb = (unsigned short*)(ws + 10 * MB);
    unsigned short* Wvb = (unsigned short*)(ws + 12 * MB);
    unsigned short* Wob = (unsigned short*)(ws + 14 * MB);
    unsigned short* Qb  = (unsigned short*)(ws + 16 * MB);   //  8 MB
    unsigned short* Kbb = (unsigned short*)(ws + 24 * MB);   //  8 MB
    unsigned short* Vt  = (unsigned short*)(ws + 32 * MB);   //  8 MB [bh*64+d][T]
    unsigned short* Ctx = (unsigned short*)(ws + 40 * MB);   //  8 MB

    const long nx = (long)Bc * Tc * Cc;       // 4M
    const long nw = (long)Cc * Cc;            // 1M
    f32_to_bf16_k<<<(int)((nx + 255) / 256), 256, 0, stream>>>(x,  Xb,  nx);
    f32_to_bf16_k<<<(int)((nw + 255) / 256), 256, 0, stream>>>(wq, Wqb, nw);
    f32_to_bf16_k<<<(int)((nw + 255) / 256), 256, 0, stream>>>(wk, Wkb, nw);
    f32_to_bf16_k<<<(int)((nw + 255) / 256), 256, 0, stream>>>(wv, Wvb, nw);
    f32_to_bf16_k<<<(int)((nw + 255) / 256), 256, 0, stream>>>(wo, Wob, nw);

    const int Mrows = Bc * Tc;                // 4096
    dim3 gg(Cc / 64, Mrows / 128);            // (16, 32)
    gemm_bf16_k<0><<<gg, 256, 0, stream>>>(Xb, Wqb, bq, Qb,  Mrows, Cc, Cc);
    gemm_bf16_k<0><<<gg, 256, 0, stream>>>(Xb, Wkb, bk, Kbb, Mrows, Cc, Cc);
    gemm_bf16_k<1><<<gg, 256, 0, stream>>>(Xb, Wvb, bv, Vt,  Mrows, Cc, Cc);

    float* entPtr = (float*)d_out + (long)Bc * Tc * Cc;   // last element of d_out
    zero_ent_k<<<1, 1, 0, stream>>>(entPtr);

    dim3 ga(Tc / 128, Bc * Hc);               // (16, 32)
    flash_attn_k<<<ga, 256, 0, stream>>>(Qb, Kbb, Vt, Ctx, entPtr);

    gemm_bf16_k<2><<<gg, 256, 0, stream>>>(Ctx, Wob, bo, d_out, Mrows, Cc, Cc);
}
